// MultiheadLinearAttentionLayer_77807627534821
// MI455X (gfx1250) — matrix operand order repaired and checked by reading, compiled, not run on hardware
//
#include <hip/hip_runtime.h>

// ---------------------------------------------------------------------------
// Multi-head attention, MI455X (gfx1250), wave32 + WMMA bf16 16x16x32.
//
// Shapes (fixed): B=8, S=1024, D=512, H=8, HD=H*D=4096.
// Pipeline:
//   (1) f32 -> bf16 convert of X_{q,k,v} and W_{q,k,v,o}        (memory bound)
//   (2) Q/K/V projections: batched GEMM [S,D]x[D,D] per (b,h)    (WMMA)
//   (3) fused flash-attention per (b,h,16-query tile)            (WMMA+softmax)
//   (4) output projection: [S,4096]x[4096,512] per b, f32 out    (WMMA)
// Data movement: TDM tensor_load_to_lds (6-arg form) for row-major tile
// stages (K blocks, GEMM A chunks), async global->LDS (b128) for the Q tile,
// global_prefetch for the next tile. Guarded by __has_builtin with fallbacks.
// ---------------------------------------------------------------------------

constexpr int kB  = 8;
constexpr int kS  = 1024;
constexpr int kD  = 512;
constexpr int kH  = 8;
constexpr int kHD = kH * kD;

typedef __attribute__((ext_vector_type(16))) __bf16 v16bf;
typedef __attribute__((ext_vector_type(8)))  float  v8f;

union Frag { v16bf v; uint4 q[2]; };          // 32B: one WMMA A/B fragment per lane
union Oct  { uint4 q; __bf16 e[8]; unsigned short s[8]; };

#define AS1 __attribute__((address_space(1)))
#define AS3 __attribute__((address_space(3)))

#if __has_builtin(__builtin_amdgcn_tensor_load_to_lds) && \
    __has_builtin(__builtin_amdgcn_s_wait_tensorcnt)
#define HAVE_TDM 1
#else
#define HAVE_TDM 0
#endif

#if __has_builtin(__builtin_amdgcn_global_load_async_to_lds_b128)
#define HAVE_ASYNC_LDS 1
#else
#define HAVE_ASYNC_LDS 0
#endif

__device__ __forceinline__ unsigned short f2bf_bits(float f) {
  union { float f; unsigned u; } a; a.f = f;
  unsigned r = a.u + 0x7FFFu + ((a.u >> 16) & 1u);   // round-to-nearest-even
  return (unsigned short)(r >> 16);
}
__device__ __forceinline__ __bf16 f2bf(float f) {
  union { unsigned short s; __bf16 b; } o; o.s = f2bf_bits(f);
  return o.b;
}

__device__ __forceinline__ v8f wmma_bf16(const Frag& a, const Frag& b, v8f c) {
  return __builtin_amdgcn_wmma_f32_16x16x32_bf16(
      false, a.v, false, b.v, (short)0, c, false, false);
}

typedef __attribute__((ext_vector_type(4))) int  v4i_t;
typedef __attribute__((ext_vector_type(8))) int  v8i_t;
typedef __attribute__((ext_vector_type(4))) unsigned v4u_t;

#if HAVE_ASYNC_LDS
__device__ __forceinline__ void async_copy_b128(const __bf16* g, __bf16* l) {
  __builtin_amdgcn_global_load_async_to_lds_b128(
      (AS1 v4i_t*)(AS1 void*)(void*)g,
      (AS3 v4i_t*)(AS3 void*)(void*)l, 0, 0);
}
__device__ __forceinline__ void async_wait0() {
#if __has_builtin(__builtin_amdgcn_s_wait_asynccnt)
  __builtin_amdgcn_s_wait_asynccnt(0);
#else
  asm volatile("s_wait_asynccnt 0x0" ::: "memory");
#endif
}
#endif

#if HAVE_TDM
// 2D bf16 tile load, global (row_stride_elems pitch) -> LDS, with hardware
// row padding so LDS rows stay 16B-aligned.
// pad codes per D# group1: interval 0=2 DW,...,7=256 DW; amount 0=1 DW,...
__device__ __forceinline__ void tdm_load_2d_bf16(
    const void* gsrc, void* ldst, int rows, int cols, int row_stride_elems,
    int pad_interval_code, int pad_amount_code) {
  unsigned long long ga = (unsigned long long)gsrc;
  unsigned la = (unsigned)(unsigned long long)(AS3 const void*)ldst;
  v4u_t g0;
  g0[0] = 1u;                                              // count=1, user mode
  g0[1] = la;                                              // lds_addr (bytes)
  g0[2] = (unsigned)ga;                                    // global_addr[31:0]
  g0[3] = (unsigned)((ga >> 32) & 0x1FFFFFFull) | (2u << 30);  // [56:32]|type=2
  v8i_t g1;
  g1[0] = (1 << 16)                                        // data_size = 2B
        | (1 << 20)                                        // pad_enable
        | (pad_interval_code << 22) | (pad_amount_code << 25);
  g1[1] = (cols & 0xFFFF) << 16;                           // tensor_dim0 lo16
  g1[2] = ((cols >> 16) & 0xFFFF) | ((rows & 0xFFFF) << 16); // dim0 hi, dim1 lo
  g1[3] = ((rows >> 16) & 0xFFFF) | ((cols & 0xFFFF) << 16); // dim1 hi, tile0
  g1[4] = rows & 0xFFFF;                                   // tile_dim1 (dim2=0)
  g1[5] = row_stride_elems;                                // dim0_stride lo32
  g1[6] = 0;
  g1[7] = 0;
  v4i_t z4 = {0, 0, 0, 0};
  v8i_t z8 = {0, 0, 0, 0, 0, 0, 0, 0};
  __builtin_amdgcn_tensor_load_to_lds(g0, g1, z4, z4, z8, 0);
}
__device__ __forceinline__ void tensor_wait0() {
  __builtin_amdgcn_s_wait_tensorcnt(0);
}
#endif

// ---------------------------------------------------------------- convert ---
__global__ __launch_bounds__(256) void k_cvt_bf16(
    const float* __restrict__ src, unsigned short* __restrict__ dst, int n) {
  int i = blockIdx.x * 256 + threadIdx.x;
  if (i < n) dst[i] = f2bf_bits(src[i]);
}

// ------------------------------------------------------------------- GEMM ---
// C[z] = A[z/bCount] (MxKd, row-major bf16) * Bm[z%bCount] (KdxN, row-major).
// Block: 256 threads (8 waves). Block tile 128(M) x 64(N), K-chunks of 32.
// A chunk staged by the TDM (one descriptor per chunk, HW row padding);
// B chunk transposed into LDS so B fragments are contiguous ds_load_b128.
template <bool OUT_F32>
__global__ __launch_bounds__(256) void k_gemm_bf16(
    const unsigned short* __restrict__ Ap, const unsigned short* __restrict__ Bp,
    void* __restrict__ Cout, int M, int N, int Kd, int ldA, int ldB, int ldC,
    size_t sA, size_t sB, size_t sC, int bCount) {
  __shared__ __bf16 As[128 * 40];
  __shared__ __bf16 Bs[64 * 40];

  const int z  = blockIdx.z;
  const __bf16* A  = (const __bf16*)Ap + (size_t)(z / bCount) * sA;
  const __bf16* Bm = (const __bf16*)Bp + (size_t)(z % bCount) * sB;
  const int n0 = blockIdx.x * 64;
  const int m0 = blockIdx.y * 128;
  const int t    = threadIdx.x;
  const int wave = t >> 5, lane = t & 31;
  const int ml = lane & 15, hi = lane >> 4;

  const v8f vzero = {0.f, 0.f, 0.f, 0.f, 0.f, 0.f, 0.f, 0.f};
  v8f acc[4] = {vzero, vzero, vzero, vzero};

  const int arow = t >> 1, ahalf = t & 1;     // A stage: 128 rows x 2 halves
  const int bk = t >> 3, bn = (t & 7) * 8;    // B stage: 32 k-rows x 8n groups

  for (int k0 = 0; k0 < Kd; k0 += 32) {
    // ---- stage A chunk (128 x 32): TDM descriptor, else manual copy ----
#if HAVE_TDM
    if (wave == 0) {
      // tile rows=128, cols=32 elems; pad: 32 elems = 16 DW (code 3) rows,
      // +16B (4 DW, code 3) -> LDS row stride 40 elems = 80B.
      tdm_load_2d_bf16(A + (size_t)m0 * ldA + k0, As, 128, 32, ldA, 3, 3);
      tensor_wait0();
    }
#elif HAVE_ASYNC_LDS
    {
      const __bf16* g = A + (size_t)(m0 + arow) * ldA + k0 + ahalf * 16;
      __bf16* l = As + arow * 40 + ahalf * 16;
      async_copy_b128(g, l);
      async_copy_b128(g + 8, l + 8);
    }
#else
    {
      const __bf16* g = A + (size_t)(m0 + arow) * ldA + k0 + ahalf * 16;
      uint4 d0 = ((const uint4*)g)[0];
      uint4 d1 = ((const uint4*)g)[1];
      uint4* s = (uint4*)(As + arow * 40 + ahalf * 16);
      s[0] = d0; s[1] = d1;
    }
#endif
    // ---- stage B chunk (32 x 64), transposed into Bs[n][k] ----
    {
      Oct u;
      u.q = *(const uint4*)(Bm + (size_t)(k0 + bk) * ldB + n0 + bn);
      #pragma unroll
      for (int j = 0; j < 8; ++j) Bs[(bn + j) * 40 + bk] = u.e[j];
    }
    // prefetch next chunk into GL2 while this one is consumed
    if (k0 + 32 < Kd) {
      __builtin_prefetch(A + (size_t)(m0 + arow) * ldA + k0 + 32, 0, 1);
      __builtin_prefetch(Bm + (size_t)(k0 + 32 + bk) * ldB + n0 + bn, 0, 1);
    }
#if !HAVE_TDM && HAVE_ASYNC_LDS
    async_wait0();
#endif
    __syncthreads();

    Frag af;
    const __bf16* ap = As + (wave * 16 + ml) * 40;
    af.q[0] = *(const uint4*)(ap + hi * 16);       // K = hi*16 .. +7
    af.q[1] = *(const uint4*)(ap + 8 + hi * 16);   // K = hi*16+8 .. +7
    #pragma unroll
    for (int tt = 0; tt < 4; ++tt) {
      Frag bf_;
      const __bf16* bp = Bs + (tt * 16 + ml) * 40 + hi * 16;
      bf_.q[0] = ((const uint4*)bp)[0];
      bf_.q[1] = ((const uint4*)bp)[1];
      acc[tt] = wmma_bf16(af, bf_, acc[tt]);
    }
    __syncthreads();
  }

  // epilogue: C layout = lane (r, l) -> row r + 8*(l>=16), col l&15
  if constexpr (OUT_F32) {
    float* C = (float*)Cout + (size_t)z * sC;
    #pragma unroll
    for (int tt = 0; tt < 4; ++tt)
      #pragma unroll
      for (int r = 0; r < 8; ++r)
        C[(size_t)(m0 + wave * 16 + r + hi * 8) * ldC + n0 + tt * 16 + ml] =
            acc[tt][r];
  } else {
    __bf16* C = (__bf16*)Cout + (size_t)z * sC;
    #pragma unroll
    for (int tt = 0; tt < 4; ++tt)
      #pragma unroll
      for (int r = 0; r < 8; ++r)
        C[(size_t)(m0 + wave * 16 + r + hi * 8) * ldC + n0 + tt * 16 + ml] =
            f2bf(acc[tt][r]);
  }
}

// -------------------------------------------------------- fused attention ---
// One block (128 thr = 4 waves) per (b,h, 16-query tile). Flash-style online
// softmax over 32-key blocks. Each wave redundantly computes the 16x32 score
// tile, then applies P.V only on its private 128-wide slice of D.
// K blocks staged by TDM; Q tile staged by async global->LDS; V transposed
// through registers (needs a data transpose the DMA paths can't express).
__global__ __launch_bounds__(128) void k_attn(
    const unsigned short* __restrict__ Qp, const unsigned short* __restrict__ Kp,
    const unsigned short* __restrict__ Vp, unsigned short* __restrict__ Hp) {
  __shared__ __bf16 Qs[16 * 520];     // 16 queries x 512 (stride 520, 16B rows)
  __shared__ __bf16 Ks[32 * 520];     // 32 keys x 512, row-major
  __shared__ __bf16 Vt[512 * 40];     // V block transposed: Vt[d][key]
  __shared__ __bf16 Ps[4][16 * 40];   // per-wave P tile (16x32), A-frag relayout

  const int qt = blockIdx.x, bh = blockIdx.y;
  const int b = bh >> 3, h = bh & 7;
  const int q0 = qt * 16;
  const int t = threadIdx.x, wave = t >> 5, lane = t & 31;
  const int ml = lane & 15, hi = lane >> 4;

  const __bf16* Qb = (const __bf16*)Qp + (size_t)bh * kS * kD;
  const __bf16* Kb = (const __bf16*)Kp + (size_t)bh * kS * kD;
  const __bf16* Vb = (const __bf16*)Vp + (size_t)bh * kS * kD;

  // stage the Q tile once (16 x 512)
#if HAVE_ASYNC_LDS
  {
    const int row = t >> 3, seg = (t & 7) * 64;
    const __bf16* g = Qb + (size_t)(q0 + row) * kD + seg;
    __bf16* l = Qs + row * 520 + seg;
    #pragma unroll
    for (int j = 0; j < 8; ++j) async_copy_b128(g + j * 8, l + j * 8);
  }
  async_wait0();
#else
  {
    const int row = t >> 3, seg = (t & 7) * 64;
    const uint4* g = (const uint4*)(Qb + (size_t)(q0 + row) * kD + seg);
    uint4* s = (uint4*)(Qs + row * 520 + seg);
    #pragma unroll
    for (int j = 0; j < 8; ++j) s[j] = g[j];
  }
#endif

  const v8f vzero = {0.f, 0.f, 0.f, 0.f, 0.f, 0.f, 0.f, 0.f};
  v8f o[8] = {vzero, vzero, vzero, vzero, vzero, vzero, vzero, vzero};
  float mrow[8], lrow[8];
  #pragma unroll
  for (int r = 0; r < 8; ++r) { mrow[r] = -1e30f; lrow[r] = 0.f; }

  const int krow = t >> 2, seg2 = (t & 3) * 128;
  const float sc = 0.04419417382415922f;   // 1/sqrt(512)

  for (int n0 = 0; n0 < kS; n0 += 32) {
    __syncthreads();   // previous iteration done reading Ks/Vt
    // ---- stage K block (32 x 512) row-major: TDM descriptor ----
#if HAVE_TDM
    if (wave == 0) {
      // pad: 512 elems = 256 DW rows (code 7), +16B (4 DW, code 3) -> 520
      tdm_load_2d_bf16(Kb + (size_t)n0 * kD, Ks, 32, kD, kD, 7, 3);
      tensor_wait0();
    }
#else
    {
      const uint4* g = (const uint4*)(Kb + (size_t)(n0 + krow) * kD + seg2);
      uint4* s = (uint4*)(Ks + krow * 520 + seg2);
      #pragma unroll
      for (int j = 0; j < 16; ++j) s[j] = g[j];
    }
#endif
    // ---- stage V block transposed -> Vt[d][key] ----
    {
      const uint4* g = (const uint4*)(Vb + (size_t)(n0 + krow) * kD + seg2);
      #pragma unroll
      for (int j = 0; j < 16; ++j) {
        Oct u; u.q = g[j];
        #pragma unroll
        for (int e = 0; e < 8; ++e) Vt[(seg2 + j * 8 + e) * 40 + krow] = u.e[e];
      }
    }
    // prefetch next key block while this one is consumed
    if (n0 + 32 < kS) {
      __builtin_prefetch(Kb + (size_t)(n0 + 32 + krow) * kD + seg2, 0, 1);
      __builtin_prefetch(Vb + (size_t)(n0 + 32 + krow) * kD + seg2, 0, 1);
    }
    __syncthreads();

    // S = Q.K^T : two 16x16 f32 tiles (keys n0..+15, n0+16..+31)
    v8f s0 = vzero, s1 = vzero;
    for (int d0 = 0; d0 < kD; d0 += 32) {
      Frag af, b0, b1;
      const __bf16* ap = Qs + ml * 520 + d0;
      af.q[0] = *(const uint4*)(ap + hi * 16);
      af.q[1] = *(const uint4*)(ap + 8 + hi * 16);
      const __bf16* p0 = Ks + ml * 520 + d0 + hi * 16;         // key = ml
      b0.q[0] = ((const uint4*)p0)[0];
      b0.q[1] = ((const uint4*)p0)[1];
      const __bf16* p1 = Ks + (16 + ml) * 520 + d0 + hi * 16;  // key = 16+ml
      b1.q[0] = ((const uint4*)p1)[0];
      b1.q[1] = ((const uint4*)p1)[1];
      s0 = wmma_bf16(af, b0, s0);
      s1 = wmma_bf16(af, b1, s1);
    }

    // online softmax; N is striped over 16 lanes per half -> xor-butterfly
    #pragma unroll
    for (int r = 0; r < 8; ++r) {
      float a0 = s0[r] * sc, a1 = s1[r] * sc;
      float v = fmaxf(a0, a1);
      #pragma unroll
      for (int off = 8; off >= 1; off >>= 1) v = fmaxf(v, __shfl_xor(v, off, 32));
      float mn = fmaxf(mrow[r], v);
      float alpha = __expf(mrow[r] - mn);
      mrow[r] = mn;
      float e0 = __expf(a0 - mn), e1 = __expf(a1 - mn);
      float rs = e0 + e1;
      #pragma unroll
      for (int off = 8; off >= 1; off >>= 1) rs += __shfl_xor(rs, off, 32);
      lrow[r] = lrow[r] * alpha + rs;
      #pragma unroll
      for (int tt = 0; tt < 8; ++tt) o[tt][r] *= alpha;
      __bf16* pr = &Ps[wave][(r + hi * 8) * 40];   // row-major 16x32 P
      pr[ml]      = f2bf(e0);
      pr[16 + ml] = f2bf(e1);
    }

    // O += P.V on this wave's 128-wide D slice (DS ops in-order per wave)
    {
      Frag af;
      const __bf16* pp = &Ps[wave][ml * 40];
      af.q[0] = *(const uint4*)(pp + hi * 16);
      af.q[1] = *(const uint4*)(pp + 8 + hi * 16);
      const int dbase = wave * 128;
      #pragma unroll
      for (int tt = 0; tt < 8; ++tt) {
        Frag bf_;
        const __bf16* vp = Vt + (dbase + tt * 16 + ml) * 40 + hi * 16;
        bf_.q[0] = ((const uint4*)vp)[0];
        bf_.q[1] = ((const uint4*)vp)[1];
        o[tt] = wmma_bf16(af, bf_, o[tt]);
      }
    }
  }

  // normalize by softmax denominator, write heads[b][q][h*512 + slice]
  __bf16* Hb = (__bf16*)Hp + ((size_t)b * kS + q0) * kHD + h * kD + wave * 128;
  #pragma unroll
  for (int r = 0; r < 8; ++r) {
    float inv = 1.0f / lrow[r];
    const int row = r + hi * 8;
    #pragma unroll
    for (int tt = 0; tt < 8; ++tt)
      Hb[(size_t)row * kHD + tt * 16 + ml] = f2bf(o[tt][r] * inv);
  }
}

// ------------------------------------------------------------------- host ---
extern "C" void kernel_launch(void* const* d_in, const int* in_sizes, int n_in,
                              void* d_out, int out_size, void* d_ws, size_t ws_size,
                              hipStream_t stream) {
  const float* Xq = (const float*)d_in[0];
  const float* Xk = (const float*)d_in[1];
  const float* Xv = (const float*)d_in[2];
  const float* Wq = (const float*)d_in[3];
  const float* Wk = (const float*)d_in[4];
  const float* Wv = (const float*)d_in[5];
  const float* Wo = (const float*)d_in[6];

  char* ws = (char*)d_ws;
  size_t off = 0;
  auto take = [&](size_t bytes) -> unsigned short* {
    unsigned short* p = (unsigned short*)(ws + off);
    off += (bytes + 255) & ~(size_t)255;
    return p;
  };

  const size_t nX = (size_t)kB * kS * kD;        // per X tensor
  const size_t nW = (size_t)kH * kD * kD;        // per weight tensor
  const size_t nQ = (size_t)kB * kH * kS * kD;   // Q/K/V
  const size_t nHd = (size_t)kB * kS * kHD;      // heads [B,S,H*D]

  unsigned short* Xqb = take(nX * 2);
  unsigned short* Xkb = take(nX * 2);
  unsigned short* Xvb = take(nX * 2);
  unsigned short* Wqb = take(nW * 2);
  unsigned short* Wkb = take(nW * 2);
  unsigned short* Wvb = take(nW * 2);
  unsigned short* Wob = take(nW * 2);
  unsigned short* Qb  = take(nQ * 2);
  unsigned short* Kb  = take(nQ * 2);
  unsigned short* Vb  = take(nQ * 2);
  unsigned short* Hd  = take(nHd * 2);

  // (1) convert inputs + weights to bf16
  k_cvt_bf16<<<dim3((unsigned)((nX + 255) / 256)), 256, 0, stream>>>(Xq, Xqb, (int)nX);
  k_cvt_bf16<<<dim3((unsigned)((nX + 255) / 256)), 256, 0, stream>>>(Xk, Xkb, (int)nX);
  k_cvt_bf16<<<dim3((unsigned)((nX + 255) / 256)), 256, 0, stream>>>(Xv, Xvb, (int)nX);
  k_cvt_bf16<<<dim3((unsigned)((nW + 255) / 256)), 256, 0, stream>>>(Wq, Wqb, (int)nW);
  k_cvt_bf16<<<dim3((unsigned)((nW + 255) / 256)), 256, 0, stream>>>(Wk, Wkb, (int)nW);
  k_cvt_bf16<<<dim3((unsigned)((nW + 255) / 256)), 256, 0, stream>>>(Wv, Wvb, (int)nW);
  k_cvt_bf16<<<dim3((unsigned)((nW + 255) / 256)), 256, 0, stream>>>(Wo, Wob, (int)nW);

  // (2) projections: z = b*H + h ; A = X[b], B = W[h], C = (Q|K|V)[b,h]
  dim3 gp(kD / 64, kS / 128, kB * kH);
  k_gemm_bf16<false><<<gp, 256, 0, stream>>>(Xqb, Wqb, Qb, kS, kD, kD, kD, kD, kD,
      (size_t)kS * kD, (size_t)kD * kD, (size_t)kS * kD, kH);
  k_gemm_bf16<false><<<gp, 256, 0, stream>>>(Xkb, Wkb, Kb, kS, kD, kD, kD, kD, kD,
      (size_t)kS * kD, (size_t)kD * kD, (size_t)kS * kD, kH);
  k_gemm_bf16<false><<<gp, 256, 0, stream>>>(Xvb, Wvb, Vb, kS, kD, kD, kD, kD, kD,
      (size_t)kS * kD, (size_t)kD * kD, (size_t)kS * kD, kH);

  // (3) fused attention -> heads [B,S,H*D] (bf16)
  dim3 ga(kS / 16, kB * kH);
  k_attn<<<ga, 128, 0, stream>>>(Qb, Kb, Vb, Hd);

  // (4) output projection: per b, [S,4096] x [4096,512] -> f32 d_out
  dim3 go(kD / 64, kS / 128, kB);
  k_gemm_bf16<true><<<go, 256, 0, stream>>>(Hd, Wob, d_out, kS, kD, kHD, kHD, kD, kD,
      (size_t)kS * kHD, (size_t)0, (size_t)kS * kD, 1);
}
